// NodeClsGAT_7533372637724
// MI455X (gfx1250) — matrix operand order repaired and checked by reading, not compiled, not run on hardware
//
#include <hip/hip_runtime.h>

typedef __attribute__((ext_vector_type(16))) __bf16 v16bf;
typedef __attribute__((ext_vector_type(8)))  float  v8f;

#define NEG_SLOPE 0.2f

// ---------------------------------------------------------------------------
// Zero a float buffer, float4-vectorized (grid-stride). Scalar tail included.
__global__ void k_zero_f32(float* __restrict__ p, long long count) {
    long long i = (long long)blockIdx.x * blockDim.x + threadIdx.x;
    long long stride = (long long)gridDim.x * blockDim.x;
    long long count4 = count >> 2;
    float4* p4 = (float4*)p;
    const float4 z = make_float4(0.f, 0.f, 0.f, 0.f);
    for (long long j = i; j < count4; j += stride) p4[j] = z;
    long long tail = count4 << 2;
    for (long long j = tail + i; j < count; j += stride) p[j] = 0.0f;
}

// ---------------------------------------------------------------------------
// Pre-pack W[FI,FO] (f32, row-major) into bf16 B-fragments in a contiguous-K WMMA
// 32x16 layout: fragment (nt,ks) = column tile nt, K-step ks. Per lane:
// n = lane&15, K-half = lane>>4; 16 bf16 = 32 bytes, coalesced & aligned.
// Columns >= FO are zeroed here so the GEMM hot loop is branch-free.
template <int FI, int FO>
__global__ void k_prep_wfrag(const float* __restrict__ W,
                             __bf16* __restrict__ Wfrag) {
    constexpr int KSTEPS   = FI / 32;
    constexpr int FO_TILES = (FO + 15) / 16;
    const int lane = threadIdx.x & 31;
    const int wid  = (int)((blockIdx.x * blockDim.x + threadIdx.x) >> 5);
    if (wid >= FO_TILES * KSTEPS) return;
    const int nt = wid / KSTEPS;
    const int ks = wid - nt * KSTEPS;

    const int m    = lane & 15;
    const int half = lane >> 4;
    const int nn   = nt * 16 + m;
    const bool nok = nn < FO;
    const int nsafe = nok ? nn : 0;

    __bf16* dst = Wfrag + ((size_t)wid * 32 + lane) * 16;
#pragma unroll
    for (int p = 0; p < 8; ++p) {
        int k = ks * 32 + half * 16 + 2 * p;
        float w0 = W[k * FO + nsafe];
        float w1 = W[(k + 1) * FO + nsafe];
        dst[2 * p]     = (__bf16)(nok ? w0 : 0.0f);
        dst[2 * p + 1] = (__bf16)(nok ? w1 : 0.0f);
    }
}

// ---------------------------------------------------------------------------
// H[n,FO] = X[n,FI] @ W[FI,FO] via v_wmma_f32_16x16x32_bf16.
// One wave per 16-row tile, fully unrolled. A fragments built once, reused
// across column tiles; B fragments are pre-packed aligned v16bf loads.
// Store epilogue: single wave-uniform branch picks the unguarded fast path.
template <int FI, int FO>
__global__ void k_gemm_wmma_bf16(const float* __restrict__ X,
                                 const __bf16* __restrict__ Wfrag,
                                 float* __restrict__ H,
                                 int n) {
    constexpr int KSTEPS   = FI / 32;
    constexpr int FO_TILES = (FO + 15) / 16;

    const int lane = threadIdx.x & 31;
    const int wid  = (int)(((long long)blockIdx.x * blockDim.x + threadIdx.x) >> 5);
    if (wid * 16 >= n) return;                 // wave-uniform exit (EXEC stays full)

    const int m    = lane & 15;                // A row / B,D column index
    const int half = lane >> 4;                // K-half selector (contiguous K: k = 16*half + s)
    int row = wid * 16 + m;
    if (row >= n) row = n - 1;                 // safe read; row not stored
    const float* xr = X + (long long)row * FI;

    // A fragments: contiguous-K 16x32 layout used here (lanes 0-15: K 0-15;
    // lanes 16-31: K 16-31), 2 packed K per VGPR slot.
    v16bf afrag[KSTEPS];
#pragma unroll
    for (int ks = 0; ks < KSTEPS; ++ks) {
#pragma unroll
        for (int p = 0; p < 8; ++p) {
            int kb = half * 16 + 2 * p;
            int k  = ks * 32 + kb;
            afrag[ks][2 * p]     = (__bf16)xr[k];
            afrag[ks][2 * p + 1] = (__bf16)xr[k + 1];
        }
    }

    const bool fullTile = (wid * 16 + 16 <= n);     // wave-uniform
    const v16bf* Wf = (const v16bf*)Wfrag;          // 32B-aligned fragments
#pragma unroll
    for (int nt = 0; nt < FO_TILES; ++nt) {
        v8f acc = {};
#pragma unroll
        for (int ks = 0; ks < KSTEPS; ++ks) {
            v16bf b = Wf[(nt * KSTEPS + ks) * 32 + lane];
            acc = __builtin_amdgcn_wmma_f32_16x16x32_bf16(
                false, afrag[ks], false, b, (short)0, acc, false, false);
        }
        // D layout: vgpr r -> row (half*8 + r), col = lane&15.
        const int nn = nt * 16 + m;
        const unsigned base = (unsigned)(wid * 16 + half * 8) * FO + (unsigned)nn;
        if (fullTile) {
            if (nn < FO) {
#pragma unroll
                for (int r = 0; r < 8; ++r)
                    H[base + (unsigned)(r * FO)] = acc[r];
            }
        } else {
            if (nn < FO) {
#pragma unroll
                for (int r = 0; r < 8; ++r) {
                    int mrow = wid * 16 + half * 8 + r;
                    if (mrow < n) H[base + (unsigned)(r * FO)] = acc[r];
                }
            }
        }
    }
}

// ---------------------------------------------------------------------------
// Per-node attention logits: as[i] = h[i,:].a_s ; ad[i] = h[i,:].a_d
// One wave per node, shuffle reduction (wave32).
template <int FO>
__global__ void k_alpha(const float* __restrict__ h,
                        const float* __restrict__ a_s,
                        const float* __restrict__ a_d,
                        float* __restrict__ as, float* __restrict__ ad,
                        int n) {
    const int lane = threadIdx.x & 31;
    const int wid  = (int)(((long long)blockIdx.x * blockDim.x + threadIdx.x) >> 5);
    if (wid >= n) return;
    float ss = 0.0f, sd = 0.0f;
#pragma unroll
    for (int f0 = 0; f0 < FO; f0 += 32) {
        int f = f0 + lane;
        if (f < FO) {
            float hv = h[(long long)wid * FO + f];
            ss += hv * a_s[f];
            sd += hv * a_d[f];
        }
    }
#pragma unroll
    for (int o = 16; o > 0; o >>= 1) {
        ss += __shfl_down(ss, o, 32);
        sd += __shfl_down(sd, o, 32);
    }
    if (lane == 0) { as[wid] = ss; ad[wid] = sd; }
}

// ---------------------------------------------------------------------------
// Edge pass 1: score -> leaky_relu -> exp; accumulate softmax denominator.
// Edges [0,E) from edge_index, [E, E+n) are self loops.
__global__ void k_edge_score(const int* __restrict__ src,
                             const int* __restrict__ dst,
                             const float* __restrict__ as,
                             const float* __restrict__ ad,
                             float* __restrict__ ew,
                             float* __restrict__ denom,
                             int E, int n) {
    const int i = blockIdx.x * blockDim.x + threadIdx.x;
    const int total = E + n;
    if (i >= total) return;
    int s, d;
    if (i < E) {
        if (i + 16384 < E) {                 // stream the edge list ahead of use
            __builtin_prefetch(&src[i + 16384], 0, 0);
            __builtin_prefetch(&dst[i + 16384], 0, 0);
        }
        s = src[i]; d = dst[i];
    } else {
        s = d = i - E;
    }
    float e = as[s] + ad[d];
    e = (e > 0.0f) ? e : NEG_SLOPE * e;
    // exp(e)/sum exp(e) == exp(e-m)/sum exp(e-m): max-subtraction pass elided.
    float ex = __expf(e);
    ew[i] = ex;
    atomicAdd(denom + d, ex);
}

// ---------------------------------------------------------------------------
// Edge pass 2: agg[dst,:] += (ew/denom[dst]) * h[src,:]. One wave per edge;
// lanes cover the feature dimension (coalesced gather, f32 atomic scatter).
template <int FO>
__global__ void k_edge_agg(const int* __restrict__ src,
                           const int* __restrict__ dst,
                           const float* __restrict__ h,
                           const float* __restrict__ ew,
                           const float* __restrict__ denom,
                           float* __restrict__ agg,
                           int E, int n) {
    const int lane = threadIdx.x & 31;
    const long long wid = ((long long)blockIdx.x * blockDim.x + threadIdx.x) >> 5;
    const long long total = (long long)E + n;
    if (wid >= total) return;
    int s, d;
    if (wid < E) { s = src[wid]; d = dst[wid]; }
    else         { s = d = (int)(wid - E); }
    const float w = ew[wid] / denom[d];
    const float* hs = h + (long long)s * FO;
    float* od = agg + (long long)d * FO;
#pragma unroll
    for (int f0 = 0; f0 < FO; f0 += 32) {
        int f = f0 + lane;
        if (f < FO) atomicAdd(od + f, w * hs[f]);
    }
}

// ---------------------------------------------------------------------------
// y = relu(y + b) over float4 lanes (count must be a multiple of 4; FO%4==0).
template <int FO>
__global__ void k_bias_relu(float* __restrict__ y, const float* __restrict__ b,
                            long long count) {
    constexpr int FO4 = FO / 4;
    long long i = (long long)blockIdx.x * blockDim.x + threadIdx.x;
    if (i >= (count >> 2)) return;
    const float4* b4 = (const float4*)b;
    float4* y4 = (float4*)y;
    float4 v = y4[i];
    float4 bb = b4[(int)(i % FO4)];
    v.x = fmaxf(v.x + bb.x, 0.0f);
    v.y = fmaxf(v.y + bb.y, 0.0f);
    v.z = fmaxf(v.z + bb.z, 0.0f);
    v.w = fmaxf(v.w + bb.w, 0.0f);
    y4[i] = v;
}

// ---------------------------------------------------------------------------
// Final head: out[i,:] = log_softmax(agg[i,:] + b).
template <int FO>
__global__ void k_bias_logsoftmax(const float* __restrict__ agg,
                                  const float* __restrict__ b,
                                  float* __restrict__ out, int n) {
    const int i = blockIdx.x * blockDim.x + threadIdx.x;
    if (i >= n) return;
    float v[FO];
    float m = -3.402823466e38f;
#pragma unroll
    for (int f = 0; f < FO; ++f) {
        v[f] = agg[(long long)i * FO + f] + b[f];
        m = fmaxf(m, v[f]);
    }
    float ssum = 0.0f;
#pragma unroll
    for (int f = 0; f < FO; ++f) ssum += __expf(v[f] - m);
    float lse = m + __logf(ssum);
#pragma unroll
    for (int f = 0; f < FO; ++f) out[(long long)i * FO + f] = v[f] - lse;
}

// ---------------------------------------------------------------------------
extern "C" void kernel_launch(void* const* d_in, const int* in_sizes, int n_in,
                              void* d_out, int out_size, void* d_ws, size_t ws_size,
                              hipStream_t stream) {
    const int IN = 64, Hdim = 64, OUT = 10;
    const int n = in_sizes[0] / IN;          // 100000
    const int E = in_sizes[1] / 2;           // 1600000

    const float* x   = (const float*)d_in[0];
    const int*   ei  = (const int*)d_in[1];
    const int*   src = ei;
    const int*   dst = ei + E;

    const float* W[3]  = { (const float*)d_in[2],  (const float*)d_in[6],  (const float*)d_in[10] };
    const float* As[3] = { (const float*)d_in[3],  (const float*)d_in[7],  (const float*)d_in[11] };
    const float* Ad[3] = { (const float*)d_in[4],  (const float*)d_in[8],  (const float*)d_in[12] };
    const float* B[3]  = { (const float*)d_in[5],  (const float*)d_in[9],  (const float*)d_in[13] };

    // Workspace carve-up (floats): h | y | as | ad | denom | ew | wfrag
    float* ws    = (float*)d_ws;
    float* hbuf  = ws;
    float* ybuf  = hbuf + (size_t)n * Hdim;
    float* asbuf = ybuf + (size_t)n * Hdim;
    float* adbuf = asbuf + n;
    float* denom = adbuf + n;
    float* ewbuf = denom + n;
    __bf16* wfrag = (__bf16*)(ewbuf + (size_t)E + n);   // <= 8KB, 32B-aligned

    float* out = (float*)d_out;
    const long long total = (long long)E + n;

    const dim3 blk(256);
    const unsigned gemm_grid  = (unsigned)(((size_t)((n + 15) / 16) * 32 + 255) / 256);
    const unsigned alpha_grid = (unsigned)(((long long)n * 32 + 255) / 256);
    const unsigned edge_grid  = (unsigned)((total + 255) / 256);
    const unsigned agg_grid   = (unsigned)((total * 32 + 255) / 256);

    for (int L = 0; L < 3; ++L) {
        const float* xin = (L == 0) ? x : ybuf;
        const bool last = (L == 2);
        const int fo = last ? OUT : Hdim;

        // 0) pre-pack W into bf16 WMMA B-fragments (tiny)
        if (!last) k_prep_wfrag<64, 64><<<dim3(1), blk, 0, stream>>>(W[L], wfrag);
        else       k_prep_wfrag<64, 10><<<dim3(1), blk, 0, stream>>>(W[L], wfrag);

        // 1) h = xin @ W   (WMMA bf16)
        if (!last) k_gemm_wmma_bf16<64, 64><<<dim3(gemm_grid), blk, 0, stream>>>(xin, wfrag, hbuf, n);
        else       k_gemm_wmma_bf16<64, 10><<<dim3(gemm_grid), blk, 0, stream>>>(xin, wfrag, hbuf, n);

        // 2) per-node alpha_src / alpha_dst
        if (!last) k_alpha<64><<<dim3(alpha_grid), blk, 0, stream>>>(hbuf, As[L], Ad[L], asbuf, adbuf, n);
        else       k_alpha<10><<<dim3(alpha_grid), blk, 0, stream>>>(hbuf, As[L], Ad[L], asbuf, adbuf, n);

        // 3) zero denom + aggregation buffer (gemm already consumed ybuf)
        k_zero_f32<<<dim3(512), blk, 0, stream>>>(denom, (long long)n);
        k_zero_f32<<<dim3(4096), blk, 0, stream>>>(ybuf, (long long)n * fo);

        // 4) edge scores + softmax denominators
        k_edge_score<<<dim3(edge_grid), blk, 0, stream>>>(src, dst, asbuf, adbuf, ewbuf, denom, E, n);

        // 5) weighted scatter-add aggregation
        if (!last) k_edge_agg<64><<<dim3(agg_grid), blk, 0, stream>>>(src, dst, hbuf, ewbuf, denom, ybuf, E, n);
        else       k_edge_agg<10><<<dim3(agg_grid), blk, 0, stream>>>(src, dst, hbuf, ewbuf, denom, ybuf, E, n);

        // 6) epilogue
        if (!last) {
            long long count = (long long)n * Hdim;
            k_bias_relu<64><<<dim3((unsigned)(((count >> 2) + 255) / 256)), blk, 0, stream>>>(ybuf, B[L], count);
        } else {
            k_bias_logsoftmax<10><<<dim3((n + 255) / 256), blk, 0, stream>>>(ybuf, B[L], out, n);
        }
    }
    (void)n_in; (void)out_size; (void)ws_size;
}
